// Model_22170621182488
// MI455X (gfx1250) — matrix operand order repaired and checked by reading, not compiled, not run on hardware
//
#include <hip/hip_runtime.h>
#include <hip/hip_bf16.h>

// ---------------------------------------------------------------------------
// Autoformer forward for MI455X (gfx1250, wave32, WMMA).
// mean_corr is computed as diagonal sums of Q@K^T (per batch) -> pure WMMA,
// no FFT needed. All GEMMs use v_wmma_f32_16x16x32_f16 (f16 in, f32 acc).
// Score kernel streams K tiles with double-buffered GLOBAL_LOAD_ASYNC_TO_LDS.
// ---------------------------------------------------------------------------

typedef __attribute__((ext_vector_type(16))) _Float16 v16h;
typedef __attribute__((ext_vector_type(8)))  _Float16 v8h;
typedef __attribute__((ext_vector_type(8)))  float    v8f;

#define AF_B    128
#define AF_L    1024
#define AF_D    64
#define AF_DFF  128
#define AF_C    7
#define AF_TK   20

// ---------------------------------------------------------------------------
// Generic WMMA GEMM: Y[R,N] = X[R,K] @ W^T  (+bias)(+gelu)(+res). W is [N,K]
// (torch layout). Block = 128 threads = 4 waves; each wave owns 16 rows.
// ---------------------------------------------------------------------------
template<int K, int N, bool GELU, bool OUT_HALF>
__global__ __launch_bounds__(128)
void wmma_gemm(const float* __restrict__ X, const float* __restrict__ W,
               const float* __restrict__ bias, const float* __restrict__ res,
               void* __restrict__ Yv, int R)
{
    __shared__ __align__(16) _Float16 Ws[N][K];
    __shared__ __align__(16) _Float16 Xs[64][K];
    const int tid = threadIdx.x;
    const int row0 = blockIdx.x * 64;

    for (int i = tid; i < N * K; i += 128)
        (&Ws[0][0])[i] = (_Float16)W[i];
    for (int i = tid; i < 64 * K; i += 128) {
        int r = i / K, k = i % K;
        (&Xs[0][0])[i] = (_Float16)X[(size_t)(row0 + r) * K + k];
    }
    __syncthreads();

    const int wave = tid >> 5, lane = tid & 31;
    const int m = lane & 15, hi = lane >> 4;
    const int wrow = wave * 16;

    // A fragments (16x32 f16, contiguous k order used here, same as B: lanes 0-15 -> K 0..15, 16-31 -> K 16..31)
    v16h a[K / 32];
#pragma unroll
    for (int kt = 0; kt < K / 32; ++kt) {
        const int kb = kt * 32;
        a[kt] = *(const v16h*)&Xs[wrow + m][kb + hi * 16];
    }

#pragma unroll
    for (int nt = 0; nt < N / 16; ++nt) {
        const int n0 = nt * 16;
        v8f c = {};
#pragma unroll
        for (int kt = 0; kt < K / 32; ++kt) {
            const int kb = kt * 32;
            // B fragment (32x16): lane n holds col n; lanes 0-15 K 0..15, 16-31 K 16..31
            v16h bf = *(const v16h*)&Ws[n0 + m][kb + hi * 16];
            c = __builtin_amdgcn_wmma_f32_16x16x32_f16(false, a[kt], false, bf,
                                                       (short)0, c, false, false);
        }
        const int col = n0 + m;
        const float bv = bias ? bias[col] : 0.0f;
#pragma unroll
        for (int r = 0; r < 8; ++r) {
            const int rowl = wrow + r + hi * 8;        // C/D layout: lanes16-31 -> M=r+8
            const size_t g = (size_t)(row0 + rowl) * N + col;
            float val = c[r] + bv;
            if (GELU) val = 0.5f * val * (1.0f + erff(val * 0.70710678118654752f));
            if (res)  val += res[g];
            if (OUT_HALF) ((_Float16*)Yv)[g] = (_Float16)val;
            else          ((float*)Yv)[g]    = val;
        }
    }
}

// ---------------------------------------------------------------------------
// mean_corr[b,tau] += (1/64) * sum over (t'-t mod L == tau) of (Q K^T)[t',t]
// Grid: (B, L/64). Q,K are f16 [B,L,64].
// K tiles (64 rows x 64ch = 8KB) streamed via double-buffered async-to-LDS.
// ---------------------------------------------------------------------------
__global__ __launch_bounds__(128)
void score_kernel(const _Float16* __restrict__ Q, const _Float16* __restrict__ Km,
                  float* __restrict__ mean_corr)
{
    __shared__ __align__(16) _Float16 Ks[2][64][64];   // 16 KB double buffer
    __shared__ __align__(16) _Float16 Qs[64][64];
    __shared__ float diag[AF_L];

    const int b = blockIdx.x, rb = blockIdx.y;
    const int tid = threadIdx.x;
    const int row0 = rb * 64;
    const _Float16* Qb = Q  + (size_t)b * AF_L * 64;
    const _Float16* Kb = Km + (size_t)b * AF_L * 64;

    for (int i = tid; i < AF_L; i += 128) diag[i] = 0.0f;
    for (int i = tid; i < 64 * 64 / 8; i += 128)
        ((uint4*)Qs)[i] = ((const uint4*)(Qb + (size_t)row0 * 64))[i];

    const unsigned ks_base = (unsigned)(uintptr_t)&Ks[0][0][0];

    // Issue phase 0 (K rows 0..63) into LDS buffer 0 (async, ASYNCcnt-tracked).
    {
        const uint64_t ga0 = (uint64_t)(uintptr_t)Kb;
#pragma unroll
        for (int p = 0; p < 4; ++p) {
            const unsigned la = ks_base + (unsigned)(tid + p * 128) * 16u;
            const uint64_t ga = ga0 + (uint64_t)(tid + p * 128) * 16u;
            asm volatile("global_load_async_to_lds_b128 %0, %1, off"
                         :: "v"(la), "v"(ga) : "memory");
        }
    }

    __syncthreads();   // Qs ready

    const int wave = tid >> 5, lane = tid & 31;
    const int m = lane & 15, hi = lane >> 4;
    const int wrow = wave * 16;

    v16h a[2];
#pragma unroll
    for (int kt = 0; kt < 2; ++kt) {
        const int kb = kt * 32;
        a[kt] = *(const v16h*)&Qs[wrow + m][kb + hi * 16];
    }

    for (int ph = 0; ph < AF_L / 64; ++ph) {
        const int buf = ph & 1;
        if (ph < AF_L / 64 - 1) {
            // Prefetch next 64 K-rows into the other buffer while we compute.
            const uint64_t ga0 = (uint64_t)(uintptr_t)(Kb + (size_t)(ph + 1) * 64 * 64);
            const unsigned lb = ks_base + (unsigned)((ph + 1) & 1) * 8192u;
#pragma unroll
            for (int p = 0; p < 4; ++p) {
                const unsigned la = lb + (unsigned)(tid + p * 128) * 16u;
                const uint64_t ga = ga0 + (uint64_t)(tid + p * 128) * 16u;
                asm volatile("global_load_async_to_lds_b128 %0, %1, off"
                             :: "v"(la), "v"(ga) : "memory");
            }
            // Async loads complete in order: <=4 outstanding means phase ph landed.
            asm volatile("s_wait_asynccnt 4" ::: "memory");
        } else {
            asm volatile("s_wait_asynccnt 0" ::: "memory");
        }
        __syncthreads();   // all waves' tile data visible

#pragma unroll
        for (int st = 0; st < 4; ++st) {
            v8f c = {};
#pragma unroll
            for (int kt = 0; kt < 2; ++kt) {
                v16h bf = *(const v16h*)&Ks[buf][st * 16 + m][kt * 32 + hi * 16];
                c = __builtin_amdgcn_wmma_f32_16x16x32_f16(false, a[kt], false, bf,
                                                           (short)0, c, false, false);
            }
            const int tcol = ph * 64 + st * 16 + m;
#pragma unroll
            for (int r = 0; r < 8; ++r) {
                const int trow = row0 + wrow + r + hi * 8;
                const int tau = (trow - tcol) & (AF_L - 1);
                atomicAdd(&diag[tau], c[r]);
            }
        }
        __syncthreads();   // reads of this buffer done before it is refilled
    }

    float* mc = mean_corr + (size_t)b * AF_L;
    for (int i = tid; i < AF_L; i += 128)
        atomicAdd(&mc[i], diag[i] * (1.0f / 64.0f));
}

// ---------------------------------------------------------------------------
// Top-k(20) + softmax per batch. One block per b.
// ---------------------------------------------------------------------------
__global__ __launch_bounds__(256)
void topk_kernel(const float* __restrict__ mean_corr,
                 float* __restrict__ wout, int* __restrict__ dout)
{
    __shared__ float vals[AF_L];
    __shared__ float rv[256];
    __shared__ int   ri[256];
    __shared__ float selv[AF_TK];
    __shared__ int   seli[AF_TK];
    const int b = blockIdx.x, tid = threadIdx.x;

    for (int i = tid; i < AF_L; i += 256) vals[i] = mean_corr[(size_t)b * AF_L + i];
    __syncthreads();

    for (int it = 0; it < AF_TK; ++it) {
        float bv = -3.4e38f; int bi = 0;
        for (int i = tid; i < AF_L; i += 256)
            if (vals[i] > bv) { bv = vals[i]; bi = i; }
        rv[tid] = bv; ri[tid] = bi;
        __syncthreads();
        for (int s = 128; s > 0; s >>= 1) {
            if (tid < s && rv[tid + s] > rv[tid]) { rv[tid] = rv[tid + s]; ri[tid] = ri[tid + s]; }
            __syncthreads();
        }
        if (tid == 0) { selv[it] = rv[0]; seli[it] = ri[0]; vals[ri[0]] = -3.4e38f; }
        __syncthreads();
    }
    if (tid == 0) {
        float mx = selv[0];
        for (int i = 1; i < AF_TK; ++i) mx = fmaxf(mx, selv[i]);
        float s = 0.0f, e[AF_TK];
        for (int i = 0; i < AF_TK; ++i) { e[i] = expf(selv[i] - mx); s += e[i]; }
        const float inv = 1.0f / s;
        for (int i = 0; i < AF_TK; ++i) {
            wout[b * AF_TK + i] = e[i] * inv;
            dout[b * AF_TK + i] = seli[i];
        }
    }
}

// agg[b,l,d] = sum_i w_i * v[b,(l+delay_i)%L,d]
__global__ void aggregate_kernel(const float* __restrict__ V, const float* __restrict__ Wgt,
                                 const int* __restrict__ Dly, float* __restrict__ Out)
{
    const size_t idx = (size_t)blockIdx.x * 256 + threadIdx.x;
    if (idx >= (size_t)AF_B * AF_L * AF_D) return;
    const int d = idx & 63;
    const size_t bl = idx >> 6;
    const int l = bl & (AF_L - 1);
    const int b = (int)(bl >> 10);
    float acc = 0.0f;
#pragma unroll
    for (int i = 0; i < AF_TK; ++i) {
        const int dl = Dly[b * AF_TK + i];
        acc += Wgt[b * AF_TK + i] *
               V[((size_t)b * AF_L + ((l + dl) & (AF_L - 1))) * AF_D + d];
    }
    Out[idx] = acc;
}

// moving_avg k=25 with replicate padding; optional detrend out and ma store/accum.
__global__ void movavg_kernel(const float* __restrict__ X, float* __restrict__ detr,
                              float* __restrict__ maout, int C, int accumulate)
{
    const size_t idx = (size_t)blockIdx.x * 256 + threadIdx.x;
    if (idx >= (size_t)AF_B * AF_L * C) return;
    const int c = idx % C;
    const int l = (int)((idx / C) % AF_L);
    const int b = (int)(idx / ((size_t)C * AF_L));
    float s = 0.0f;
#pragma unroll 5
    for (int j = -12; j <= 12; ++j) {
        int ll = l + j; ll = ll < 0 ? 0 : (ll > AF_L - 1 ? AF_L - 1 : ll);
        s += X[((size_t)b * AF_L + ll) * C + c];
    }
    const float ma = s * (1.0f / 25.0f);
    if (detr)  detr[idx] = X[idx] - ma;
    if (maout) { if (accumulate) maout[idx] += ma; else maout[idx] = ma; }
}

// mean over L: [B,L,C] -> [B,C]
__global__ void meanL_kernel(const float* __restrict__ X, float* __restrict__ M, int C)
{
    const int idx = blockIdx.x * 256 + threadIdx.x;
    if (idx >= AF_B * C) return;
    const int b = idx / C, c = idx % C;
    float s = 0.0f;
    for (int l = 0; l < AF_L; ++l) s += X[((size_t)b * AF_L + l) * C + c];
    M[idx] = s * (1.0f / AF_L);
}

// trend_init / seasonal_init construction (decoder positions 0..1023)
__global__ void build_init_kernel(const float* __restrict__ xenc, const float* __restrict__ ma7,
                                  const float* __restrict__ mean7,
                                  float* __restrict__ trinit, float* __restrict__ sfull)
{
    const size_t idx = (size_t)blockIdx.x * 256 + threadIdx.x;
    if (idx >= (size_t)AF_B * AF_L * AF_C) return;
    const int c = idx % AF_C;
    const int l = (int)((idx / AF_C) % AF_L);
    const int b = (int)(idx / ((size_t)AF_C * AF_L));
    if (l < 512) {
        const size_t s = ((size_t)b * AF_L + (512 + l)) * AF_C + c;
        trinit[idx] = ma7[s];
        sfull[idx]  = xenc[s] - ma7[s];
    } else {
        trinit[idx] = mean7[b * AF_C + c];
        sfull[idx]  = 0.0f;
    }
}

// Encoder embedding: circular conv (Cin=7,k=3) + temporal table lookups
__global__ __launch_bounds__(64)
void embed_enc_kernel(const float* __restrict__ x, const float* __restrict__ convw,
                      const float* __restrict__ emb, const int* __restrict__ mark,
                      float* __restrict__ out)
{
    const int bl = blockIdx.x;
    const int b = bl >> 10, l = bl & (AF_L - 1);
    const int o = threadIdx.x;
    __shared__ float xr[3][AF_C];
    __shared__ int mk[4];
    if (o < 21) {
        const int k = o / 7, c = o % 7;
        const int ll = (l - 1 + k + AF_L) & (AF_L - 1);
        xr[k][c] = x[((size_t)b * AF_L + ll) * AF_C + c];
    }
    if (o < 4) mk[o] = mark[((size_t)b * AF_L + l) * 4 + o];
    __syncthreads();
    float acc = 0.0f;
#pragma unroll
    for (int k = 0; k < 3; ++k)
#pragma unroll
        for (int c = 0; c < 7; ++c) acc += xr[k][c] * convw[(o * 7 + c) * 3 + k];
#pragma unroll
    for (int f = 0; f < 4; ++f) acc += emb[(f * 32 + mk[f]) * 64 + o];
    out[((size_t)b * AF_L + l) * 64 + o] = acc;
}

// Decoder embedding: circular conv + x_mark_dec @ dec_time_W
__global__ __launch_bounds__(64)
void embed_dec_kernel(const float* __restrict__ x, const float* __restrict__ convw,
                      const float* __restrict__ xmark, const float* __restrict__ timeW,
                      float* __restrict__ out)
{
    const int bl = blockIdx.x;
    const int b = bl >> 10, l = bl & (AF_L - 1);
    const int o = threadIdx.x;
    __shared__ float xr[3][AF_C];
    __shared__ float xm[4];
    if (o < 21) {
        const int k = o / 7, c = o % 7;
        const int ll = (l - 1 + k + AF_L) & (AF_L - 1);
        xr[k][c] = x[((size_t)b * AF_L + ll) * AF_C + c];
    }
    if (o < 4) xm[o] = xmark[((size_t)b * AF_L + l) * 4 + o];
    __syncthreads();
    float acc = 0.0f;
#pragma unroll
    for (int k = 0; k < 3; ++k)
#pragma unroll
        for (int c = 0; c < 7; ++c) acc += xr[k][c] * convw[(o * 7 + c) * 3 + k];
#pragma unroll
    for (int f = 0; f < 4; ++f) acc += xm[f] * timeW[f * 64 + o];
    out[((size_t)b * AF_L + l) * 64 + o] = acc;
}

// LayerNorm over last dim (D=64), one thread per (b,l)
__global__ void ln_kernel(const float* __restrict__ X, const float* __restrict__ w,
                          const float* __restrict__ bias, float* __restrict__ Y)
{
    const size_t bl = (size_t)blockIdx.x * 256 + threadIdx.x;
    if (bl >= (size_t)AF_B * AF_L) return;
    const float* x = X + bl * AF_D;
    float mu = 0.0f;
#pragma unroll
    for (int d = 0; d < AF_D; ++d) mu += x[d];
    mu *= (1.0f / AF_D);
    float var = 0.0f;
#pragma unroll
    for (int d = 0; d < AF_D; ++d) { const float t = x[d] - mu; var += t * t; }
    var *= (1.0f / AF_D);
    const float inv = rsqrtf(var + 1e-5f);
    float* y = Y + bl * AF_D;
#pragma unroll
    for (int d = 0; d < AF_D; ++d) y[d] = (x[d] - mu) * inv * w[d] + bias[d];
}

// y = x - seq_mean[b,d]
__global__ void sub_mean_kernel(const float* __restrict__ X, const float* __restrict__ M,
                                float* __restrict__ Y)
{
    const size_t idx = (size_t)blockIdx.x * 256 + threadIdx.x;
    if (idx >= (size_t)AF_B * AF_L * AF_D) return;
    const int d = idx & 63;
    const int b = (int)(idx >> 16);           // idx / (L*D) = idx / 65536
    Y[idx] = X[idx] - M[b * AF_D + d];
}

// out[b,lo,c] = trend_init + circ_conv3(tacc, trendw) + x_final @ proj_W + proj_b
__global__ void output_kernel(const float* __restrict__ trinit, const float* __restrict__ tacc,
                              const float* __restrict__ trendw, const float* __restrict__ xfin,
                              const float* __restrict__ projW, const float* __restrict__ projb,
                              float* __restrict__ out)
{
    const int idx = blockIdx.x * 256 + threadIdx.x;
    if (idx >= AF_B * 512 * AF_C) return;
    const int c = idx % AF_C;
    const int lo = (idx / AF_C) % 512;
    const int b = idx / (AF_C * 512);
    const int l = 512 + lo;
    float acc = trinit[((size_t)b * AF_L + l) * AF_C + c] + projb[c];
#pragma unroll
    for (int k = 0; k < 3; ++k) {
        const int ll = (l - 1 + k) & (AF_L - 1);
        const float* t = tacc + ((size_t)b * AF_L + ll) * AF_D;
        float s = 0.0f;
        for (int d = 0; d < AF_D; ++d) s += t[d] * trendw[(c * AF_D + d) * 3 + k];
        acc += s;
    }
    const float* xf = xfin + ((size_t)b * AF_L + l) * AF_D;
    for (int d = 0; d < AF_D; ++d) acc += xf[d] * projW[d * AF_C + c];
    out[idx] = acc;
}

// ---------------------------------------------------------------------------
extern "C" void kernel_launch(void* const* d_in, const int* in_sizes, int n_in,
                              void* d_out, int out_size, void* d_ws, size_t ws_size,
                              hipStream_t stream)
{
    const float* x_enc      = (const float*)d_in[0];
    const float* x_mark_dec = (const float*)d_in[2];
    const float* enc_conv_w = (const float*)d_in[3];
    const float* emb_tables = (const float*)d_in[4];
    const float* dec_conv_w = (const float*)d_in[5];
    const float* dec_time_W = (const float*)d_in[6];
    const float* enc_attn_W = (const float*)d_in[7];
    const float* enc_attn_b = (const float*)d_in[8];
    const float* enc_ffn_W1 = (const float*)d_in[9];
    const float* enc_ffn_W2 = (const float*)d_in[10];
    const float* enc_norm_w = (const float*)d_in[11];
    const float* enc_norm_b = (const float*)d_in[12];
    const float* dec_self_W = (const float*)d_in[13];
    const float* dec_self_b = (const float*)d_in[14];
    const float* dec_cross_W= (const float*)d_in[15];
    const float* dec_cross_b= (const float*)d_in[16];
    const float* dec_ffn_W1 = (const float*)d_in[17];
    const float* dec_ffn_W2 = (const float*)d_in[18];
    const float* dec_trend_w= (const float*)d_in[19];
    const float* dec_norm_w = (const float*)d_in[20];
    const float* dec_norm_b = (const float*)d_in[21];
    const float* proj_W     = (const float*)d_in[22];
    const float* proj_b     = (const float*)d_in[23];
    const int*   x_mark_enc = (const int*)d_in[24];

    const size_t BL = (size_t)AF_B * AF_L;

    char* wp = (char*)d_ws;
    auto alloc = [&](size_t bytes) -> void* {
        void* p = wp; wp += (bytes + 255) & ~(size_t)255; return p;
    };
    float*    X    = (float*)alloc(BL * AF_D * 4);
    float*    T1   = (float*)alloc(BL * AF_D * 4);
    float*    T2   = (float*)alloc(BL * AF_D * 4);
    float*    Vb   = (float*)alloc(BL * AF_D * 4);
    float*    TACC = (float*)alloc(BL * AF_D * 4);
    float*    ENC  = (float*)alloc(BL * AF_D * 4);
    float*    HID  = (float*)alloc(BL * AF_DFF * 4);
    _Float16* QH   = (_Float16*)alloc(BL * AF_D * 2);
    _Float16* KH   = (_Float16*)alloc(BL * AF_D * 2);
    float*    MA7  = (float*)alloc(BL * AF_C * 4);
    float*    TRIN = (float*)alloc(BL * AF_C * 4);
    float*    SFUL = (float*)alloc(BL * AF_C * 4);
    float*    MEAN7= (float*)alloc(AF_B * AF_C * 4);
    float*    MCORR= (float*)alloc(BL * 4);
    float*    MCOL = (float*)alloc(AF_B * AF_D * 4);
    float*    WGT  = (float*)alloc(AF_B * AF_TK * 4);
    int*      DLY  = (int*)alloc(AF_B * AF_TK * 4);

    const int gemm_blocks = (int)(BL / 64);          // 2048
    const int bld_blocks  = (int)((BL * AF_D + 255) / 256);
    const int blc_blocks  = (int)((BL * AF_C + 255) / 256);

    auto run_attn = [&](const float* Xq, const float* Xkv, const float* Wset,
                        const float* bset, float* outbuf) {
        wmma_gemm<64,64,false,true ><<<gemm_blocks,128,0,stream>>>(Xq,  Wset + 0*4096, bset + 0,   nullptr, QH, (int)BL);
        wmma_gemm<64,64,false,true ><<<gemm_blocks,128,0,stream>>>(Xkv, Wset + 1*4096, bset + 64,  nullptr, KH, (int)BL);
        wmma_gemm<64,64,false,false><<<gemm_blocks,128,0,stream>>>(Xkv, Wset + 2*4096, bset + 128, nullptr, Vb, (int)BL);
        hipMemsetAsync(MCORR, 0, BL * 4, stream);
        score_kernel<<<dim3(AF_B, AF_L/64),128,0,stream>>>(QH, KH, MCORR);
        topk_kernel<<<AF_B,256,0,stream>>>(MCORR, WGT, DLY);
        aggregate_kernel<<<bld_blocks,256,0,stream>>>(Vb, WGT, DLY, T1);
        wmma_gemm<64,64,false,false><<<gemm_blocks,128,0,stream>>>(T1, Wset + 3*4096, bset + 192, Xq, outbuf, (int)BL);
    };
    auto run_ffn = [&](const float* Xin, const float* W1, const float* W2, float* outbuf) {
        wmma_gemm<64,128,true ,false><<<gemm_blocks,128,0,stream>>>(Xin, W1, nullptr, nullptr, HID, (int)BL);
        wmma_gemm<128,64,false,false><<<gemm_blocks,128,0,stream>>>(HID, W2, nullptr, Xin, outbuf, (int)BL);
    };

    // ---- encoder-input decomposition --------------------------------------
    movavg_kernel<<<blc_blocks,256,0,stream>>>(x_enc, nullptr, MA7, AF_C, 0);
    meanL_kernel<<<(AF_B*AF_C + 255)/256,256,0,stream>>>(x_enc, MEAN7, AF_C);
    build_init_kernel<<<blc_blocks,256,0,stream>>>(x_enc, MA7, MEAN7, TRIN, SFUL);

    // ---- encoder ----------------------------------------------------------
    embed_enc_kernel<<<(int)BL,64,0,stream>>>(x_enc, enc_conv_w, emb_tables, x_mark_enc, X);
    for (int l = 0; l < 2; ++l) {
        run_attn(X, X, enc_attn_W + (size_t)l*4*4096, enc_attn_b + (size_t)l*256, T2);
        movavg_kernel<<<bld_blocks,256,0,stream>>>(T2, X, nullptr, AF_D, 0);
        run_ffn(X, enc_ffn_W1 + (size_t)l*AF_DFF*AF_D, enc_ffn_W2 + (size_t)l*AF_D*AF_DFF, T1);
        movavg_kernel<<<bld_blocks,256,0,stream>>>(T1, X, nullptr, AF_D, 0);
    }
    ln_kernel<<<(int)((BL + 255)/256),256,0,stream>>>(X, enc_norm_w, enc_norm_b, T1);
    meanL_kernel<<<(AF_B*AF_D + 255)/256,256,0,stream>>>(T1, MCOL, AF_D);
    sub_mean_kernel<<<bld_blocks,256,0,stream>>>(T1, MCOL, ENC);

    // ---- decoder ----------------------------------------------------------
    embed_dec_kernel<<<(int)BL,64,0,stream>>>(SFUL, dec_conv_w, x_mark_dec, dec_time_W, X);
    hipMemsetAsync(TACC, 0, BL * AF_D * 4, stream);

    run_attn(X, X, dec_self_W, dec_self_b, T2);
    movavg_kernel<<<bld_blocks,256,0,stream>>>(T2, X, TACC, AF_D, 1);

    run_attn(X, ENC, dec_cross_W, dec_cross_b, T2);
    movavg_kernel<<<bld_blocks,256,0,stream>>>(T2, X, TACC, AF_D, 1);

    run_ffn(X, dec_ffn_W1, dec_ffn_W2, T1);
    movavg_kernel<<<bld_blocks,256,0,stream>>>(T1, X, TACC, AF_D, 1);

    ln_kernel<<<(int)((BL + 255)/256),256,0,stream>>>(X, dec_norm_w, dec_norm_b, T1);
    meanL_kernel<<<(AF_B*AF_D + 255)/256,256,0,stream>>>(T1, MCOL, AF_D);
    sub_mean_kernel<<<bld_blocks,256,0,stream>>>(T1, MCOL, T2);

    output_kernel<<<(AF_B*512*AF_C + 255)/256,256,0,stream>>>(
        TRIN, TACC, dec_trend_w, T2, proj_W, proj_b, (float*)d_out);
}
